// Informer_73083163508982
// MI455X (gfx1250) — matrix operand order repaired and checked by reading, not compiled, not run on hardware
//
#include <hip/hip_runtime.h>
#include <cstdint>

// ============================================================================
// Informer forward for MI455X (gfx1250, wave32, WMMA).
// All matmuls run through v_wmma_f32_16x16x32_bf16 (bf16 in, f32 accum).
// Every GEMM operand is kept row-major with K contiguous so both the A and B
// WMMA fragments are per-lane contiguous 16B loads (no shuffles, no LDS):
//   A frag (16x32 bf16): lane m=lane&15, kb=(lane>>4)*8, elems = K[kb..kb+7],
//                        K[kb+16..kb+23]                      (ISA 7.12.2)
//   B frag (32x16 bf16): lane n=lane&15, kb=(lane>>4)*8, elems = K[kb..kb+7],
//                        K[kb+16..kb+23]
//   C/D (16x16 f32): lane<16 -> n=lane, rows m0..m0+7; lane>=16 -> rows +8.
// GEMM v2: 2x2 register tile per wave (32x32 out), branchless ping-pong
// double-buffered K loop (requires K % 64 == 0, M % 32 == 0, Npad % 32 == 0).
// ============================================================================

typedef __attribute__((ext_vector_type(16))) __bf16 v16bf;
typedef __attribute__((ext_vector_type(8)))  float  v8f;

#define BATCH 16
#define LSEQ  384
#define CIN   2356
#define DM    512
#define NH    8
#define HD    64
#define DFF   2048
#define BL    (BATCH*LSEQ)          // 6144
#define KCONV_REAL 7068             // 3*CIN
#define KCONV 7104                  // padded to multiple of 64 (ping-pong loop)
#define NPROJ 2368                  // CIN padded to multiple of 32
#define UTOP  30                    // 5*ceil(ln(384)) = 30
#define NSAMP 30
#define LN_EPS 1e-5f

__device__ __forceinline__ unsigned short f2bf(float f){
  unsigned int u = __float_as_uint(f);
  u += 0x7FFFu + ((u >> 16) & 1u);          // round-to-nearest-even
  return (unsigned short)(u >> 16);
}

// ---------------------------------------------------------------------------
// Generic batched WMMA GEMM:  C = A[M,K] * B[N,K]^T  (+bias, optional relu)
// batch z -> (outer=z/Hin, inner=z%Hin), per-matrix two-level batch strides.
// Requirements (guaranteed by host code): K%64==0, M%32==0, B rows padded to
// a multiple of 32 (zero rows), every launched N-block fully inside Npad.
// ---------------------------------------------------------------------------
struct GemmP {
  const unsigned short* A;  long long sAo, sAi;  int lda;
  const unsigned short* Bm; long long sBo, sBi;  int ldb;
  float*          C;        long long sCo, sCi;  int ldc;
  unsigned short* Cb;       long long sCbo, sCbi;int ldcb;
  const float* bias;
  int M, N, K, Hin, flags;  // flags bit0 = relu
};

union FragBF { v16bf v; uint4 q[2]; };
struct Frags { FragBF a0, a1, b0, b1; };

__device__ __forceinline__ void loadFrags(Frags& f,
    const unsigned short* arow0, const unsigned short* arow1,
    const unsigned short* brow0, const unsigned short* brow1, int k){
  const uint4* ap0 = (const uint4*)(arow0 + k);
  f.a0.q[0] = ap0[0]; f.a0.q[1] = ap0[2];        // A: K split +0 / +16
  const uint4* ap1 = (const uint4*)(arow1 + k);
  f.a1.q[0] = ap1[0]; f.a1.q[1] = ap1[2];
  const uint4* bp0 = (const uint4*)(brow0 + k);
  f.b0.q[0] = bp0[0]; f.b0.q[1] = bp0[2];        // B: K split +0 / +16
  const uint4* bp1 = (const uint4*)(brow1 + k);
  f.b1.q[0] = bp1[0]; f.b1.q[1] = bp1[2];
}

#define WMMA_BF16(A_, B_, C_) \
  __builtin_amdgcn_wmma_f32_16x16x32_bf16(false, (A_), false, (B_), (short)0, (C_), false, false)

__global__ __launch_bounds__(256) void k_gemm(GemmP p){
  const int z  = blockIdx.z;
  const int bo = z / p.Hin, bi = z % p.Hin;
  const unsigned short* Abase = p.A  + bo*p.sAo + bi*p.sAi;
  const unsigned short* Bbase = p.Bm + bo*p.sBo + bi*p.sBi;
  const int tilesN = (p.N + 63) >> 6;
  const int bm = blockIdx.x / tilesN;
  const int bn = blockIdx.x % tilesN;
  const int wave = threadIdx.x >> 5;
  const int lane = threadIdx.x & 31;
  const int m0 = bm*128 + (wave & 3)*32;
  const int n0 = bn*64  + (wave >> 2)*32;
  const int Npad = (p.N + 31) & ~31;
  if (m0 >= p.M || n0 >= Npad) return;           // uniform per wave
  const int r     = lane & 15;
  const int kaOff = (lane >> 4) * 8;             // A fragment K base
  const int kbOff = (lane >> 4) * 8;             // B fragment K base
  const unsigned short* arow0 = Abase + (size_t)(m0 + r)      * p.lda + kaOff;
  const unsigned short* arow1 = Abase + (size_t)(m0 + 16 + r) * p.lda + kaOff;
  const unsigned short* brow0 = Bbase + (size_t)(n0 + r)      * p.ldb + kbOff;
  const unsigned short* brow1 = Bbase + (size_t)(n0 + 16 + r) * p.ldb + kbOff;

  v8f c00 = {0.f,0.f,0.f,0.f,0.f,0.f,0.f,0.f};
  v8f c01 = c00, c10 = c00, c11 = c00;
  Frags f0, f1;
  loadFrags(f0, arow0, arow1, brow0, brow1, 0);
  for (int k = 0; k < p.K; k += 64){
    loadFrags(f1, arow0, arow1, brow0, brow1, k + 32);
    c00 = WMMA_BF16(f0.a0.v, f0.b0.v, c00);
    c01 = WMMA_BF16(f0.a0.v, f0.b1.v, c01);
    c10 = WMMA_BF16(f0.a1.v, f0.b0.v, c10);
    c11 = WMMA_BF16(f0.a1.v, f0.b1.v, c11);
    if (k + 64 < p.K) loadFrags(f0, arow0, arow1, brow0, brow1, k + 64);
    c00 = WMMA_BF16(f1.a0.v, f1.b0.v, c00);
    c01 = WMMA_BF16(f1.a0.v, f1.b1.v, c01);
    c10 = WMMA_BF16(f1.a1.v, f1.b0.v, c10);
    c11 = WMMA_BF16(f1.a1.v, f1.b1.v, c11);
  }

  float*          Cf = p.C  ? (p.C  + bo*p.sCo  + bi*p.sCi ) : nullptr;
  unsigned short* Cb = p.Cb ? (p.Cb + bo*p.sCbo + bi*p.sCbi) : nullptr;
  const int half8 = (lane >> 4) * 8;
  #pragma unroll
  for (int mt = 0; mt < 2; ++mt){
    #pragma unroll
    for (int nt = 0; nt < 2; ++nt){
      v8f acc = (mt == 0) ? (nt == 0 ? c00 : c01) : (nt == 0 ? c10 : c11);
      const int col = n0 + nt*16 + r;
      if (col >= p.N) continue;
      const int mbase = m0 + mt*16 + half8;
      const float bv = p.bias ? p.bias[col] : 0.f;
      #pragma unroll
      for (int i = 0; i < 8; ++i){
        float v = acc[i] + bv;
        if (p.flags & 1) v = fmaxf(v, 0.f);
        const size_t row = (size_t)(mbase + i);
        if (Cf) Cf[row * p.ldc  + col] = v;
        if (Cb) Cb[row * p.ldcb + col] = f2bf(v);
      }
    }
  }
}

// ---------------------------------------------------------------------------
// LayerNorm with generic addressing (supports the strided dec_norm over L),
// optional fused residual add, writes f32 and/or bf16.
// element offset = (r/innerCount)*outerStride + (r%innerCount)*innerStride
//                + j*elemStride
// ---------------------------------------------------------------------------
struct LnP {
  const float* x; const float* add; const float* g; const float* bv;
  float* out; unsigned short* outb;
  int C; int innerCount; long long innerStride, outerStride, elemStride;
};

__global__ __launch_bounds__(256) void k_ln(LnP p){
  __shared__ float sm[256];
  const int rIdx = blockIdx.x;
  const long long base = (long long)(rIdx / p.innerCount) * p.outerStride
                       + (long long)(rIdx % p.innerCount) * p.innerStride;
  const int t = threadIdx.x;
  float s = 0.f;
  for (int j = t; j < p.C; j += 256){
    const long long o = base + (long long)j * p.elemStride;
    float v = p.x[o]; if (p.add) v += p.add[o];
    s += v;
  }
  sm[t] = s; __syncthreads();
  for (int d = 128; d > 0; d >>= 1){ if (t < d) sm[t] += sm[t+d]; __syncthreads(); }
  const float mean = sm[0] / (float)p.C; __syncthreads();
  float vs = 0.f;
  for (int j = t; j < p.C; j += 256){
    const long long o = base + (long long)j * p.elemStride;
    float v = p.x[o]; if (p.add) v += p.add[o];
    v -= mean; vs += v * v;
  }
  sm[t] = vs; __syncthreads();
  for (int d = 128; d > 0; d >>= 1){ if (t < d) sm[t] += sm[t+d]; __syncthreads(); }
  const float rstd = rsqrtf(sm[0] / (float)p.C + LN_EPS); __syncthreads();
  for (int j = t; j < p.C; j += 256){
    const long long o = base + (long long)j * p.elemStride;
    float v = p.x[o]; if (p.add) v += p.add[o];
    const float y = (v - mean) * rstd * p.g[j] + p.bv[j];
    if (p.out)  p.out[o]  = y;
    if (p.outb) p.outb[o] = f2bf(y);
  }
}

// softmax over rows of length n (stride ld); applies scale; writes bf16
__global__ __launch_bounds__(128)
void k_softmax(const float* S, unsigned short* P, int ld, int n, float scale){
  __shared__ float sm[128];
  const float* row = S + (size_t)blockIdx.x * ld;
  unsigned short* pr = P + (size_t)blockIdx.x * ld;
  const int t = threadIdx.x;
  float mx = -3e38f;
  for (int j = t; j < n; j += 128) mx = fmaxf(mx, row[j] * scale);
  sm[t] = mx; __syncthreads();
  for (int d = 64; d > 0; d >>= 1){ if (t < d) sm[t] = fmaxf(sm[t], sm[t+d]); __syncthreads(); }
  const float m = sm[0]; __syncthreads();
  float s = 0.f;
  for (int j = t; j < n; j += 128) s += __expf(row[j] * scale - m);
  sm[t] = s; __syncthreads();
  for (int d = 64; d > 0; d >>= 1){ if (t < d) sm[t] += sm[t+d]; __syncthreads(); }
  const float inv = 1.f / sm[0];
  for (int j = t; j < n; j += 128) pr[j] = f2bf(__expf(row[j] * scale - m) * inv);
}

// ---------------- ProbSparse attention helper kernels ----------------------
// sparsity score: M = max_s(q.k_idx[s]) - sum_s(q.k_idx[s]) / Lk  (hash idx)
__global__ __launch_bounds__(256)
void k_mscore(const float* Q, const float* K, float* Ms){
  const int i = blockIdx.x * blockDim.x + threadIdx.x;
  if (i >= BATCH*NH*LSEQ) return;
  const int l = i % LSEQ, h = (i / LSEQ) % NH, b = i / (LSEQ * NH);
  const float* q = Q + ((size_t)(b*LSEQ + l)) * DM + h*HD;
  float mx = -3e38f, sum = 0.f;
  for (int s = 0; s < NSAMP; ++s){
    unsigned int hsh = (unsigned)(l * 2654435761u) ^ (unsigned)(s * 2246822519u) ^ 0x9E3779B9u;
    hsh ^= hsh >> 13; hsh *= 0x85EBCA6Bu; hsh ^= hsh >> 16;
    const int idx = (int)(hsh % (unsigned)LSEQ);
    const float* kp = K + ((size_t)(b*LSEQ + idx)) * DM + h*HD;
    float d = 0.f;
    #pragma unroll 8
    for (int e = 0; e < HD; ++e) d += q[e] * kp[e];
    mx = fmaxf(mx, d); sum += d;
  }
  Ms[i] = mx - sum / (float)LSEQ;
}

// top-UTOP argmax per (b,h); iterative LDS argmax, deterministic
__global__ __launch_bounds__(512)
void k_topk(const float* Ms, int* mtop){
  __shared__ float vals[LSEQ];
  __shared__ float rv[512];
  __shared__ int   ri[512];
  const int t = threadIdx.x, bh = blockIdx.x;
  if (t < LSEQ) vals[t] = Ms[(size_t)bh * LSEQ + t];
  __syncthreads();
  for (int it = 0; it < UTOP; ++it){
    rv[t] = (t < LSEQ) ? vals[t] : -3e38f; ri[t] = t;
    __syncthreads();
    for (int d = 256; d > 0; d >>= 1){
      if (t < d){ if (rv[t+d] > rv[t]){ rv[t] = rv[t+d]; ri[t] = ri[t+d]; } }
      __syncthreads();
    }
    if (t == 0){ mtop[bh*32 + it] = ri[0]; vals[ri[0]] = -3e38f; }
    __syncthreads();
  }
}

__global__ __launch_bounds__(256)
void k_qred(const float* Q, const int* mtop, unsigned short* Qr){
  const int i = blockIdx.x * blockDim.x + threadIdx.x;
  if (i >= BATCH*NH*32*HD) return;
  const int e = i % HD, u = (i/HD) % 32, h = (i/(HD*32)) % NH, b = i/(HD*32*NH);
  float v = 0.f;
  if (u < UTOP){
    const int row = mtop[(b*NH + h)*32 + u];
    v = Q[((size_t)(b*LSEQ + row)) * DM + h*HD + e];
  }
  Qr[i] = f2bf(v);
}

// V (B,L,H,E) f32 -> Vt (B,H,E,L) bf16 (row-major K=L, for O = P * Vt^T)
__global__ __launch_bounds__(256)
void k_vt(const float* V, unsigned short* Vt){
  const long long j = (long long)blockIdx.x * blockDim.x + threadIdx.x;
  if (j >= (long long)BATCH*NH*HD*LSEQ) return;
  const int s = (int)(j % LSEQ);
  long long t = j / LSEQ;
  const int e = (int)(t % HD); t /= HD;
  const int h = (int)(t % NH); const int b = (int)(t / NH);
  Vt[j] = f2bf(V[((size_t)(b*LSEQ + s)) * DM + h*HD + e]);
}

__global__ __launch_bounds__(256)
void k_vmean(const float* V, float* vm){
  const int i = blockIdx.x * blockDim.x + threadIdx.x;
  if (i >= BATCH*NH*HD) return;
  const int e = i % HD, h = (i/HD) % NH, b = i/(HD*NH);
  float s = 0.f;
  for (int l = 0; l < LSEQ; ++l) s += V[((size_t)(b*LSEQ + l)) * DM + h*HD + e];
  vm[i] = s / (float)LSEQ;
}

__global__ __launch_bounds__(256)
void k_fill(const float* vm, float* O){
  const long long j = (long long)blockIdx.x * blockDim.x + threadIdx.x;
  if (j >= (long long)BL*DM) return;
  const int d = (int)(j % DM);
  const int b = (int)(j / ((long long)LSEQ * DM));
  O[j] = vm[(b*NH + d/HD)*HD + (d % HD)];
}

__global__ __launch_bounds__(256)
void k_scatter(const float* upd, const int* mtop, float* O){
  const int i = blockIdx.x * blockDim.x + threadIdx.x;
  if (i >= BATCH*NH*UTOP*HD) return;
  const int e = i % HD, u = (i/HD) % UTOP, h = (i/(HD*UTOP)) % NH, b = i/(HD*UTOP*NH);
  const int row = mtop[(b*NH + h)*32 + u];
  O[((size_t)(b*LSEQ + row)) * DM + h*HD + e] =
      upd[(((size_t)(b*NH + h))*32 + u) * HD + e];
}

// ---------------- misc conversion / layout kernels -------------------------
__global__ __launch_bounds__(256)
void k_cvt(const float* s, unsigned short* d, long long n){
  const long long j = (long long)blockIdx.x * blockDim.x + threadIdx.x;
  if (j < n) d[j] = f2bf(s[j]);
}

// src f32 [R,C] -> dst bf16 [Rp,Cp] zero-padded
__global__ __launch_bounds__(256)
void k_cvt2d(const float* src, unsigned short* dst, int R, int C, int Cp, int Rp){
  const long long j = (long long)blockIdx.x * blockDim.x + threadIdx.x;
  if (j >= (long long)Rp * Cp) return;
  const int c = (int)(j % Cp), r = (int)(j / Cp);
  const float v = (r < R && c < C) ? src[(size_t)r * C + c] : 0.f;
  dst[j] = f2bf(v);
}

// x (B,CIN,L) f32 -> im2col bf16 [(b,l), c*3+t] (K padded to KCONV)
__global__ __launch_bounds__(256)
void k_im2col(const float* x, unsigned short* dst){
  const long long j = (long long)blockIdx.x * blockDim.x + threadIdx.x;
  if (j >= (long long)BL * KCONV) return;
  const int kk = (int)(j % KCONV);
  const int l  = (int)((j / KCONV) % LSEQ);
  const int b  = (int)(j / ((long long)KCONV * LSEQ));
  float v = 0.f;
  if (kk < KCONV_REAL){
    const int c = kk / 3, t = kk % 3, li = l + t - 1;
    if (li >= 0 && li < LSEQ) v = x[((size_t)b * CIN + c) * LSEQ + li];
  }
  dst[j] = f2bf(v);
}

// Cproj [(b,l), NPROJ] f32 -> out (B,CIN,L)
__global__ __launch_bounds__(256)
void k_out(const float* Cp, float* out){
  const long long j = (long long)blockIdx.x * blockDim.x + threadIdx.x;
  if (j >= (long long)BATCH*CIN*LSEQ) return;
  const int l = (int)(j % LSEQ);
  const long long t = j / LSEQ;
  const int c = (int)(t % CIN), b = (int)(t / CIN);
  out[j] = Cp[((size_t)(b*LSEQ + l)) * NPROJ + c];
}

// ============================================================================
// Host orchestration
// ============================================================================
extern "C" void kernel_launch(void* const* d_in, const int* in_sizes, int n_in,
                              void* d_out, int out_size, void* d_ws, size_t ws_size,
                              hipStream_t stream){
  (void)in_sizes; (void)n_in; (void)out_size; (void)ws_size;
  // Input order assumption: [0]=x_enc, [1]=x_dec, then params flattened with
  // dicts sorted alphabetically, lists in order:
  //   dec[0..1] (cross{k,o,q,v}{b,w}, ff1{b,w}, ff2{b,w}, n1{b,g}, n2{b,g},
  //              n3{b,g}, self{k,o,q,v}{b,w}) -> 26 each, base 2 / 28
  //   dec_norm{b=54,g=55}, embed{b=56,w=57},
  //   enc[0..2] (attn{k,o,q,v}{b,w}, ff1{b,w}, ff2{b,w}, n1{b,g}, n2{b,g})
  //              -> 16 each, base 58/74/90,  proj{b=106,w=107}
  auto IN = [&](int i)->const float*{ return (const float*)d_in[i]; };
  const float* x_enc = IN(0);
  const float* x_dec = IN(1);

  char* wsb = (char*)d_ws;
  size_t off = 0;
  auto alloc = [&](size_t bytes)->void*{
    void* p = wsb + off;
    off = (off + bytes + 255) & ~(size_t)255;
    return p;
  };
  auto cvtW = [&](const float* src, int R, int C, int Cp)->unsigned short*{
    const int Rp = (R + 31) & ~31;              // rows padded to mult of 32
    unsigned short* dst = (unsigned short*)alloc((size_t)Rp * Cp * 2);
    const long long n = (long long)Rp * Cp;
    k_cvt2d<<<(unsigned)((n + 255) / 256), 256, 0, stream>>>(src, dst, R, C, Cp, Rp);
    return dst;
  };

  struct AttnW { const unsigned short *q,*k,*v,*o; const float *bq,*bk,*bv,*bo; };
  auto mkAttn = [&](int base)->AttnW{
    AttnW w;
    w.bk = IN(base+0); w.k = cvtW(IN(base+1), DM, DM, DM);
    w.bo = IN(base+2); w.o = cvtW(IN(base+3), DM, DM, DM);
    w.bq = IN(base+4); w.q = cvtW(IN(base+5), DM, DM, DM);
    w.bv = IN(base+6); w.v = cvtW(IN(base+7), DM, DM, DM);
    return w;
  };

  // ---- weight prep (all bf16, padded) ----
  unsigned short* WembP = cvtW(IN(57), DM, KCONV_REAL, KCONV);
  const float* embB = IN(56);

  AttnW encAt[3]; const unsigned short *encW1[3], *encW2[3];
  const float *encB1[3], *encB2[3], *encN1g[3], *encN1b[3], *encN2g[3], *encN2b[3];
  for (int i = 0; i < 3; ++i){
    const int base = 58 + 16*i;
    encAt[i] = mkAttn(base);
    encB1[i] = IN(base+8);  encW1[i] = cvtW(IN(base+9),  DFF, DM, DM);
    encB2[i] = IN(base+10); encW2[i] = cvtW(IN(base+11), DM, DFF, DFF);
    encN1b[i] = IN(base+12); encN1g[i] = IN(base+13);
    encN2b[i] = IN(base+14); encN2g[i] = IN(base+15);
  }
  AttnW decCross[2], decSelf[2]; const unsigned short *decW1[2], *decW2[2];
  const float *decB1[2], *decB2[2];
  const float *decN1g[2], *decN1b[2], *decN2g[2], *decN2b[2], *decN3g[2], *decN3b[2];
  for (int i = 0; i < 2; ++i){
    const int base = 2 + 26*i;
    decCross[i] = mkAttn(base);
    decB1[i] = IN(base+8);  decW1[i] = cvtW(IN(base+9),  DFF, DM, DM);
    decB2[i] = IN(base+10); decW2[i] = cvtW(IN(base+11), DM, DFF, DFF);
    decN1b[i] = IN(base+12); decN1g[i] = IN(base+13);
    decN2b[i] = IN(base+14); decN2g[i] = IN(base+15);
    decN3b[i] = IN(base+16); decN3g[i] = IN(base+17);
    decSelf[i] = mkAttn(base+18);
  }
  unsigned short* projW = cvtW(IN(107), CIN, DM, DM);   // rows padded to NPROJ
  const float* projB = IN(106);
  const float* dnormB = IN(54); const float* dnormG = IN(55);

  // ---- activations / temporaries ----
  float*          eF = (float*)alloc((size_t)BL*DM*4);
  unsigned short* eB = (unsigned short*)alloc((size_t)BL*DM*2);
  float*          dF = (float*)alloc((size_t)BL*DM*4);
  unsigned short* dB = (unsigned short*)alloc((size_t)BL*DM*2);
  // union region: im2col (87.3MB) -> cross-attn S (75.5MB) -> Cproj (58.2MB)
  void* uni = alloc((size_t)BL*KCONV*2);
  unsigned short* Xcol  = (unsigned short*)uni;
  float*          Sbig  = (float*)uni;
  float*          Cproj = (float*)uni;
  unsigned short* Pbig = (unsigned short*)alloc((size_t)BATCH*NH*LSEQ*LSEQ*2);
  float* Qf = (float*)alloc((size_t)BL*DM*4);
  float* Kf = (float*)alloc((size_t)BL*DM*4);
  float* Vf = (float*)alloc((size_t)BL*DM*4);
  unsigned short* Qb = (unsigned short*)alloc((size_t)BL*DM*2);
  unsigned short* Kb = (unsigned short*)alloc((size_t)BL*DM*2);
  unsigned short* Vt = (unsigned short*)alloc((size_t)BATCH*NH*HD*LSEQ*2);
  float*          Of = (float*)alloc((size_t)BL*DM*4);
  unsigned short* Ob = (unsigned short*)alloc((size_t)BL*DM*2);
  float*      tmpOut = (float*)alloc((size_t)BL*DM*4);
  float*  Ms   = (float*)alloc((size_t)BATCH*NH*LSEQ*4);
  int*    mtop = (int*)alloc((size_t)BATCH*NH*32*4);
  unsigned short* Qred = (unsigned short*)alloc((size_t)BATCH*NH*32*HD*2);
  float*          Sred = (float*)alloc((size_t)BATCH*NH*32*LSEQ*4);
  unsigned short* Pred = (unsigned short*)alloc((size_t)BATCH*NH*32*LSEQ*2);
  float*          updB = (float*)alloc((size_t)BATCH*NH*32*HD*4);
  float*         vmean = (float*)alloc((size_t)BATCH*NH*HD*4);
  unsigned short*  Hb  = (unsigned short*)alloc((size_t)BL*DFF*2);

  // ---- launch helpers ----
  auto gemm = [&](const unsigned short* A, long long sAo, long long sAi, int lda,
                  const unsigned short* Bw, long long sBo, long long sBi, int ldb,
                  float* C, long long sCo, long long sCi, int ldc,
                  unsigned short* Cb2, long long sCbo, long long sCbi, int ldcb,
                  const float* bias, int M, int N, int K, int Hin, int batches, int flags){
    GemmP p{A,sAo,sAi,lda, Bw,sBo,sBi,ldb, C,sCo,sCi,ldc, Cb2,sCbo,sCbi,ldcb,
            bias, M,N,K,Hin,flags};
    dim3 grid((unsigned)(((M+127)/128) * ((N+63)/64)), 1, (unsigned)batches);
    k_gemm<<<grid, 256, 0, stream>>>(p);
  };
  auto ln = [&](const float* x, const float* add, const float* g, const float* bv,
                float* out, unsigned short* outb, int rows, int C,
                int innerCount, long long innerStride, long long outerStride,
                long long elemStride){
    LnP p{x,add,g,bv,out,outb,C,innerCount,innerStride,outerStride,elemStride};
    k_ln<<<(unsigned)rows, 256, 0, stream>>>(p);
  };
  auto EW = [&](long long n){ return dim3((unsigned)((n + 255) / 256)); };

  // ProbSparse self-attention (hash-sampled scores, top-30 queries dense)
  auto prob_attn = [&](const unsigned short* xb, const AttnW& w, float* attnOut){
    gemm(xb,0,0,DM, w.q,0,0,DM, Qf,0,0,DM, nullptr,0,0,0, w.bq, BL,DM,DM, 1,1,0);
    gemm(xb,0,0,DM, w.k,0,0,DM, Kf,0,0,DM, Kb,0,0,DM,      w.bk, BL,DM,DM, 1,1,0);
    gemm(xb,0,0,DM, w.v,0,0,DM, Vf,0,0,DM, nullptr,0,0,0,  w.bv, BL,DM,DM, 1,1,0);
    k_vt<<<EW((long long)BATCH*NH*HD*LSEQ),256,0,stream>>>(Vf, Vt);
    k_mscore<<<EW(BATCH*NH*LSEQ),256,0,stream>>>(Qf, Kf, Ms);
    k_topk<<<BATCH*NH, 512, 0, stream>>>(Ms, mtop);
    k_qred<<<EW(BATCH*NH*32*HD),256,0,stream>>>(Qf, mtop, Qred);
    gemm(Qred,(long long)NH*32*HD,(long long)32*HD,HD,
         Kb,(long long)LSEQ*DM,(long long)HD,DM,
         Sred,(long long)NH*32*LSEQ,(long long)32*LSEQ,LSEQ,
         nullptr,0,0,0, nullptr, 32,LSEQ,HD, NH, BATCH*NH, 0);
    k_softmax<<<BATCH*NH*32, 128, 0, stream>>>(Sred, Pred, LSEQ, LSEQ, 0.125f);
    gemm(Pred,(long long)NH*32*LSEQ,(long long)32*LSEQ,LSEQ,
         Vt,(long long)NH*HD*LSEQ,(long long)HD*LSEQ,LSEQ,
         updB,(long long)NH*32*HD,(long long)32*HD,HD,
         nullptr,0,0,0, nullptr, 32,HD,LSEQ, NH, BATCH*NH, 0);
    k_vmean<<<EW(BATCH*NH*HD),256,0,stream>>>(Vf, vmean);
    k_fill<<<EW((long long)BL*DM),256,0,stream>>>(vmean, Of);
    k_scatter<<<EW(BATCH*NH*UTOP*HD),256,0,stream>>>(updB, mtop, Of);
    k_cvt<<<EW((long long)BL*DM),256,0,stream>>>(Of, Ob, (long long)BL*DM);
    gemm(Ob,0,0,DM, w.o,0,0,DM, attnOut,0,0,DM, nullptr,0,0,0, w.bo, BL,DM,DM, 1,1,0);
  };

  // Full cross-attention (batched 384x384x64 + 384x64x384 WMMA GEMMs)
  auto cross_attn = [&](const unsigned short* xqb, const unsigned short* kvb,
                        const AttnW& w, float* attnOut){
    gemm(xqb,0,0,DM, w.q,0,0,DM, nullptr,0,0,0, Qb,0,0,DM, w.bq, BL,DM,DM, 1,1,0);
    gemm(kvb,0,0,DM, w.k,0,0,DM, nullptr,0,0,0, Kb,0,0,DM, w.bk, BL,DM,DM, 1,1,0);
    gemm(kvb,0,0,DM, w.v,0,0,DM, Vf,0,0,DM, nullptr,0,0,0, w.bv, BL,DM,DM, 1,1,0);
    k_vt<<<EW((long long)BATCH*NH*HD*LSEQ),256,0,stream>>>(Vf, Vt);
    gemm(Qb,(long long)LSEQ*DM,(long long)HD,DM,
         Kb,(long long)LSEQ*DM,(long long)HD,DM,
         Sbig,(long long)NH*LSEQ*LSEQ,(long long)LSEQ*LSEQ,LSEQ,
         nullptr,0,0,0, nullptr, LSEQ,LSEQ,HD, NH, BATCH*NH, 0);
    k_softmax<<<BATCH*NH*LSEQ, 128, 0, stream>>>(Sbig, Pbig, LSEQ, LSEQ, 0.125f);
    gemm(Pbig,(long long)NH*LSEQ*LSEQ,(long long)LSEQ*LSEQ,LSEQ,
         Vt,(long long)NH*HD*LSEQ,(long long)HD*LSEQ,LSEQ,
         Of,(long long)LSEQ*DM,(long long)HD,DM,
         nullptr,0,0,0, nullptr, LSEQ,HD,LSEQ, NH, BATCH*NH, 0);
    k_cvt<<<EW((long long)BL*DM),256,0,stream>>>(Of, Ob, (long long)BL*DM);
    gemm(Ob,0,0,DM, w.o,0,0,DM, attnOut,0,0,DM, nullptr,0,0,0, w.bo, BL,DM,DM, 1,1,0);
  };

  // ===================== forward pass =====================
  // encoder embed (conv k=3 as im2col GEMM, K padded to 7104)
  k_im2col<<<EW((long long)BL*KCONV),256,0,stream>>>(x_enc, Xcol);
  gemm(Xcol,0,0,KCONV, WembP,0,0,KCONV, eF,0,0,DM, eB,0,0,DM, embB,
       BL,DM,KCONV, 1,1,0);
  for (int i = 0; i < 3; ++i){
    prob_attn(eB, encAt[i], tmpOut);
    ln(eF, tmpOut, encN1g[i], encN1b[i], eF, eB, BL, DM, BL, DM, 0, 1);
    gemm(eB,0,0,DM, encW1[i],0,0,DM, nullptr,0,0,0, Hb,0,0,DFF, encB1[i],
         BL,DFF,DM, 1,1,1);                                  // relu fused
    gemm(Hb,0,0,DFF, encW2[i],0,0,DFF, tmpOut,0,0,DM, nullptr,0,0,0, encB2[i],
         BL,DM,DFF, 1,1,0);
    ln(eF, tmpOut, encN2g[i], encN2b[i], eF, eB, BL, DM, BL, DM, 0, 1);
  }
  // decoder embed (reuses im2col region before Sbig is needed)
  k_im2col<<<EW((long long)BL*KCONV),256,0,stream>>>(x_dec, Xcol);
  gemm(Xcol,0,0,KCONV, WembP,0,0,KCONV, dF,0,0,DM, dB,0,0,DM, embB,
       BL,DM,KCONV, 1,1,0);
  for (int i = 0; i < 2; ++i){
    prob_attn(dB, decSelf[i], tmpOut);
    ln(dF, tmpOut, decN1g[i], decN1b[i], dF, dB, BL, DM, BL, DM, 0, 1);
    cross_attn(dB, eB, decCross[i], tmpOut);
    ln(dF, tmpOut, decN2g[i], decN2b[i], dF, dB, BL, DM, BL, DM, 0, 1);
    gemm(dB,0,0,DM, decW1[i],0,0,DM, nullptr,0,0,0, Hb,0,0,DFF, decB1[i],
         BL,DFF,DM, 1,1,1);
    gemm(Hb,0,0,DFF, decW2[i],0,0,DFF, tmpOut,0,0,DM, nullptr,0,0,0, decB2[i],
         BL,DM,DFF, 1,1,0);
    ln(dF, tmpOut, decN3g[i], decN3b[i], dF, dB, BL, DM, BL, DM, 0, 1);
  }
  // dec_norm: LN over the L dimension (row=(b,dm), elem stride = DM)
  ln(dF, nullptr, dnormG, dnormB, dF, dB, BATCH*DM, LSEQ,
     DM, 1, (long long)LSEQ*DM, DM);
  // projection to CIN (N padded to 2368 via zero weight rows, store guarded)
  gemm(dB,0,0,DM, projW,0,0,DM, Cproj,0,0,NPROJ, nullptr,0,0,0, projB,
       BL,CIN,DM, 1,1,0);
  // (B,L,CIN) -> (B,CIN,L)
  k_out<<<EW((long long)BATCH*CIN*LSEQ),256,0,stream>>>(Cproj, (float*)d_out);
}
